// CausalSelfAttention_55911884259367
// MI455X (gfx1250) — matrix operand order repaired and checked by reading, not compiled, not run on hardware
//
#include <hip/hip_runtime.h>
#include <stdint.h>
#include <math.h>

// ---------------------------------------------------------------------------
// Causal self-attention for MI455X (gfx1250, wave32, WMMA, async-to-LDS).
// Pipeline: fp32->bf16 convert, QKV GEMM (WMMA bf16, double-buffered async
// staging), flash attention with transposed-S trick (WMMA bf16 + in-lane fp32
// online softmax, double-buffered K/V staging, tile-level causal skipping),
// out projection GEMM. Workspace: ~92 MB of bf16 staging in d_ws.
// ---------------------------------------------------------------------------

namespace {
constexpr int Bb   = 4;
constexpr int Tt   = 2048;
constexpr int Cc   = 1024;
constexpr int Hh   = 16;
constexpr int Dk   = 64;
constexpr int Mtot = Bb * Tt;   // 8192
constexpr int Nqkv = 3 * Cc;    // 3072
constexpr int Kdim = Cc;        // 1024
}

typedef __attribute__((ext_vector_type(16))) __bf16 v16bf;
typedef __attribute__((ext_vector_type(8)))  float  v8f;

union Frag16 { v16bf v; uint4 u[2]; __bf16 h[16]; };

__device__ __forceinline__ v8f wmma_bf16(v16bf a, v16bf b, v8f c) {
  // D(f32 16x16) = A(bf16 16x32) * B(bf16 32x16) + C
  return __builtin_amdgcn_wmma_f32_16x16x32_bf16(
      false, a, false, b, (short)0, c, false, false);
}

__device__ __forceinline__ v8f vzero() {
  v8f z;
  #pragma unroll
  for (int i = 0; i < 8; ++i) z[i] = 0.0f;
  return z;
}

// Async copy 16B global -> LDS (CDNA5, ASYNCcnt-tracked).
// LDS generic address truncates to the wave-relative LDS offset (aperture rule).
__device__ __forceinline__ void async_g2l_b128(const void* gptr, void* lptr) {
  uint32_t la = (uint32_t)(size_t)lptr;
  uint64_t ga = (uint64_t)(size_t)gptr;
  asm volatile("global_load_async_to_lds_b128 %0, %1, off"
               :: "v"(la), "v"(ga) : "memory");
}
// ASYNCcnt decrements in order -> waiting to N drains everything older than
// the N most recently issued async ops (double-buffer gating).
template <int N>
__device__ __forceinline__ void async_wait() {
  if constexpr (N == 0)      asm volatile("s_wait_asynccnt 0" ::: "memory");
  else if constexpr (N == 1) asm volatile("s_wait_asynccnt 1" ::: "memory");
  else if constexpr (N == 4) asm volatile("s_wait_asynccnt 4" ::: "memory");
  else                       asm volatile("s_wait_asynccnt 0" ::: "memory");
}

// ---------------------------------------------------------------------------
// fp32 -> bf16 conversion, 8 elements / thread (n must be a multiple of 8).
// ---------------------------------------------------------------------------
__global__ void cvt_f32_to_bf16(const float* __restrict__ src,
                                __bf16* __restrict__ dst, int n) {
  int i = (blockIdx.x * blockDim.x + threadIdx.x) * 8;
  if (i >= n) return;
  float4 a = *(const float4*)(src + i);
  float4 b = *(const float4*)(src + i + 4);
  union { __bf16 h[8]; uint4 u; } o;
  o.h[0] = (__bf16)a.x; o.h[1] = (__bf16)a.y;
  o.h[2] = (__bf16)a.z; o.h[3] = (__bf16)a.w;
  o.h[4] = (__bf16)b.x; o.h[5] = (__bf16)b.y;
  o.h[6] = (__bf16)b.z; o.h[7] = (__bf16)b.w;
  *(uint4*)(dst + i) = o.u;
}

// ---------------------------------------------------------------------------
// GEMM: out[M][N] = A[M][K] * W[N][K]^T + bias (torch Linear layout).
// Block tile 128x128, 8 waves; wave tile 32(M) x 64(N) = 2x4 WMMA tiles.
// K staged 32 wide, double-buffered async-to-LDS (DMA overlaps WMMA).
// ---------------------------------------------------------------------------
template <int NDIM, bool OUT_BF16>
__global__ __launch_bounds__(256)
void gemm_bf16_tn(const __bf16* __restrict__ A,
                  const __bf16* __restrict__ W,
                  const float* __restrict__ bias,
                  void* __restrict__ outp) {
  constexpr int KT    = 32;
  constexpr int LDS_K = KT + 8;            // 40 bf16 = 80B row stride (16B mult)
  __shared__ alignas(16) __bf16 As[2][128 * LDS_K];
  __shared__ alignas(16) __bf16 Bs[2][128 * LDS_K];

  const int bm   = blockIdx.x * 128;
  const int bn   = blockIdx.y * 128;
  const int tid  = threadIdx.x;
  const int wave = tid >> 5;
  const int lane = tid & 31;
  const int wm   = wave & 3;               // 4 wave rows * 32 M
  const int wn   = wave >> 2;              // 2 wave cols * 64 N
  const int l16  = lane & 15;
  const int hi   = lane >> 4;

  v8f acc[2][4];
  #pragma unroll
  for (int mt = 0; mt < 2; ++mt)
    #pragma unroll
    for (int nt = 0; nt < 4; ++nt) acc[mt][nt] = vzero();

  // 4 async b128 per thread per stage (2 chunks x 2 matrices).
  auto stage = [&](int buf, int k0) {
    #pragma unroll
    for (int c = 0; c < 2; ++c) {
      int chunk = tid + c * 256;           // 0..511
      int row   = chunk >> 2;
      int kc    = (chunk & 3) * 8;
      async_g2l_b128(A + (size_t)(bm + row) * Kdim + k0 + kc,
                     &As[buf][row * LDS_K + kc]);
      async_g2l_b128(W + (size_t)(bn + row) * Kdim + k0 + kc,
                     &Bs[buf][row * LDS_K + kc]);
    }
  };

  constexpr int NT = Kdim / KT;            // 32 stages
  stage(0, 0);
  for (int i = 0; i < NT; ++i) {
    const int buf = i & 1;
    if (i + 1 < NT) {
      stage(buf ^ 1, (i + 1) * KT);        // issue next tile's DMA first
      async_wait<4>();                     // drain only tile i's 4 ops
    } else {
      async_wait<0>();
    }
    __syncthreads();

    // A fragments: lane row = l16; elements K {hi*8+0..7, hi*8+16..23}
    Frag16 af[2];
    #pragma unroll
    for (int mt = 0; mt < 2; ++mt) {
      const __bf16* p = &As[buf][(wm * 32 + mt * 16 + l16) * LDS_K + hi * 8];
      af[mt].u[0] = *(const uint4*)(p);
      af[mt].u[1] = *(const uint4*)(p + 16);
    }
    // B fragments: lane col = l16; elements K {hi*8+0..7, hi*8+16..23}
    Frag16 bfr[4];
    #pragma unroll
    for (int nt = 0; nt < 4; ++nt) {
      const __bf16* p = &Bs[buf][(wn * 64 + nt * 16 + l16) * LDS_K + hi * 8];
      bfr[nt].u[0] = *(const uint4*)(p);
      bfr[nt].u[1] = *(const uint4*)(p + 16);
    }
    #pragma unroll
    for (int mt = 0; mt < 2; ++mt)
      #pragma unroll
      for (int nt = 0; nt < 4; ++nt)
        acc[mt][nt] = wmma_bf16(af[mt].v, bfr[nt].v, acc[mt][nt]);
    __syncthreads();                       // all waves done with buf
  }

  // --- epilogue: C layout lane=(col l16), VGPR v -> row v + 8*hi ------------
  #pragma unroll
  for (int mt = 0; mt < 2; ++mt)
    #pragma unroll
    for (int nt = 0; nt < 4; ++nt) {
      int col = bn + wn * 64 + nt * 16 + l16;
      float bv = bias[col];
      #pragma unroll
      for (int v = 0; v < 8; ++v) {
        int row = bm + wm * 32 + mt * 16 + v + hi * 8;
        float r = acc[mt][nt][v] + bv;
        if constexpr (OUT_BF16)
          ((__bf16*)outp)[(size_t)row * NDIM + col] = (__bf16)r;
        else
          ((float*)outp)[(size_t)row * NDIM + col] = r;
      }
    }
}

// ---------------------------------------------------------------------------
// Flash attention with transposed S: grid (B*H, T/128), 8 waves x 16 q rows.
// S^T = K * Q^T  ->  C-layout puts a q column's kv values in one lane pair.
// The per-lane element set of S^T C-layout matches the A-operand layout of P
// for O = P*V, so P never round-trips through LDS. K/V tiles double-buffered.
// Tiles are classified per wave (uniform branches, EXEC stays all-ones):
//   kv0 >  qb+15 : fully masked  -> skip compute, keep barriers
//   kv0+31 <= qb : fully visible -> scale only, no compare/select
//   else         : diagonal      -> per-element causal mask (<=2 tiles/wave)
// ---------------------------------------------------------------------------
__global__ __launch_bounds__(256)
void attn_flash(const __bf16* __restrict__ qkv, __bf16* __restrict__ attout) {
  constexpr int LDK = Dk + 8;   // 72 bf16 = 144B (16B mult)
  constexpr int LDV = 32 + 8;   // 40
  __shared__ alignas(16) __bf16 Ks[2][32 * LDK];   // K tile [kv][dk]
  __shared__ alignas(16) __bf16 Vt[2][Dk * LDV];   // V tile transposed [dk][kv]

  const int bh   = blockIdx.x;
  const int b    = bh / Hh;
  const int h    = bh % Hh;
  const int q0b  = blockIdx.y * 128;
  const int tid  = threadIdx.x;
  const int wave = tid >> 5;
  const int lane = tid & 31;
  const int l16  = lane & 15;
  const int hi   = lane >> 4;
  const int qb   = q0b + wave * 16;
  const size_t rs = (size_t)Nqkv;

  // Staging coordinates: thread -> (kv row i, dk chunk cc).
  const int si = tid >> 3;            // kv row 0..31
  const int sc = (tid & 7) * 8;       // dk chunk
  auto kvptr = [&](int kv0) {
    return qkv + (size_t)(b * Tt + kv0 + si) * rs + h * Dk + sc;
  };

  // Q^T as B operand: lane = q column qb+l16; elements dk = ks*32 + {hi*8+0..7, hi*8+16..23}.
  Frag16 qf[2];
  {
    const __bf16* p = qkv + (size_t)(b * Tt + qb + l16) * rs + h * Dk;
    #pragma unroll
    for (int ks = 0; ks < 2; ++ks) {
      qf[ks].u[0] = *(const uint4*)(p + ks * 32 + hi * 8);
      qf[ks].u[1] = *(const uint4*)(p + ks * 32 + hi * 8 + 16);
    }
  }

  float mrow = -__builtin_inff();   // running max for this lane's q column
  float lrow = 0.0f;                // running sum
  v8f o[4];
  #pragma unroll
  for (int nt = 0; nt < 4; ++nt) o[nt] = vzero();

  const int ntiles = (q0b + 128) / 32;    // causal bound for this block

  // Prologue: DMA K(0), fetch V(0) into registers.
  async_g2l_b128(kvptr(0) + Cc, &Ks[0][si * LDK + sc]);
  uint4 vreg = *(const uint4*)(kvptr(0) + 2 * Cc);

  for (int t = 0; t < ntiles; ++t) {
    const int buf = t & 1;
    const int kv0 = t * 32;

    // Commit V(t) registers -> transposed LDS tile.
    {
      union { uint4 u; __bf16 hh[8]; } vv;
      vv.u = vreg;
      #pragma unroll
      for (int e = 0; e < 8; ++e) Vt[buf][(sc + e) * LDV + si] = vv.hh[e];
    }
    if (t + 1 < ntiles) {
      async_g2l_b128(kvptr((t + 1) * 32) + Cc, &Ks[buf ^ 1][si * LDK + sc]);
      vreg = *(const uint4*)(kvptr((t + 1) * 32) + 2 * Cc);  // overlaps compute
      async_wait<1>();                  // drain only K(t)'s DMA
    } else {
      async_wait<0>();
    }
    __syncthreads();                    // (compiler adds dscnt wait for Vt)

    // Wave-uniform causal classification (scalar branches; barriers outside).
    const bool tile_live = (kv0 <= qb + 15);
    const bool need_mask = (kv0 + 31 > qb);
    if (tile_live) {
      // --- S^T = K * Q^T : A = K rows (kv), B = Q^T columns (q) -------------
      v8f s[2];                         // sub covers kv sub*16..+15
      #pragma unroll
      for (int sub = 0; sub < 2; ++sub) {
        s[sub] = vzero();
        #pragma unroll
        for (int ks = 0; ks < 2; ++ks) {
          Frag16 kf;
          const __bf16* p = &Ks[buf][(sub * 16 + l16) * LDK + ks * 32 + hi * 8];
          kf.u[0] = *(const uint4*)(p);
          kf.u[1] = *(const uint4*)(p + 16);
          s[sub] = wmma_bf16(kf.v, qf[ks].v, s[sub]);
        }
      }
      // scale (+ causal mask only on diagonal tiles)
      if (need_mask) {
        #pragma unroll
        for (int sub = 0; sub < 2; ++sub)
          #pragma unroll
          for (int v = 0; v < 8; ++v) {
            int q  = qb + l16;
            int kv = kv0 + sub * 16 + v + 8 * hi;
            float val = s[sub][v] * 0.125f;      // 1/sqrt(64)
            s[sub][v] = (kv <= q) ? val : -__builtin_inff();
          }
      } else {
        #pragma unroll
        for (int sub = 0; sub < 2; ++sub)
          #pragma unroll
          for (int v = 0; v < 8; ++v) s[sub][v] *= 0.125f;
      }

      // --- online softmax: in-lane over 16 kv values + one xor16 combine ----
      float tm = -__builtin_inff();
      #pragma unroll
      for (int sub = 0; sub < 2; ++sub)
        #pragma unroll
        for (int v = 0; v < 8; ++v) tm = fmaxf(tm, s[sub][v]);
      tm = fmaxf(tm, __shfl_xor(tm, 16, 32));
      float nm = fmaxf(mrow, tm);
      float rescale = __expf(mrow - nm);         // first iter: exp(-inf)=0
      mrow = nm;

      // exp + pack P directly into A-operand layout (register-only transpose):
      // element e<8 -> kv 8*hi+e (sub0,v=e); e>=8 -> kv 16+8*hi+(e-8) (sub1).
      Frag16 pfr;
      float ts = 0.0f;
      #pragma unroll
      for (int v = 0; v < 8; ++v) {
        float p0 = __expf(s[0][v] - nm);
        float p1 = __expf(s[1][v] - nm);
        ts += p0 + p1;
        pfr.h[v]     = (__bf16)p0;
        pfr.h[8 + v] = (__bf16)p1;
      }
      ts += __shfl_xor(ts, 16, 32);
      lrow = lrow * rescale + ts;

      // broadcast rescale into C-layout rows (row q = v+8*hi lives in lane v+8*hi)
      float rf[8];
      #pragma unroll
      for (int v = 0; v < 8; ++v) rf[v] = __shfl(rescale, v + 8 * hi, 32);
      #pragma unroll
      for (int nt = 0; nt < 4; ++nt)
        #pragma unroll
        for (int v = 0; v < 8; ++v) o[nt][v] *= rf[v];

      // --- O += P * V : B = V columns (dk), kv {hi*8+0..7, hi*8+16..23} from Vt
      #pragma unroll
      for (int nt = 0; nt < 4; ++nt) {
        Frag16 vf;
        const __bf16* p = &Vt[buf][(nt * 16 + l16) * LDV + hi * 8];
        vf.u[0] = *(const uint4*)(p);
        vf.u[1] = *(const uint4*)(p + 16);
        o[nt] = wmma_bf16(pfr.v, vf.v, o[nt]);
      }
    }
    __syncthreads();                    // all waves done with buf
  }

  // --- normalize and store [B,T,H*Dk] bf16 ----------------------------------
  float linv = 1.0f / lrow;
  float iv[8];
  #pragma unroll
  for (int v = 0; v < 8; ++v) iv[v] = __shfl(linv, v + 8 * hi, 32);
  #pragma unroll
  for (int nt = 0; nt < 4; ++nt)
    #pragma unroll
    for (int v = 0; v < 8; ++v) {
      int q   = qb + v + 8 * hi;
      int col = h * Dk + nt * 16 + l16;
      attout[(size_t)(b * Tt + q) * Cc + col] = (__bf16)(o[nt][v] * iv[v]);
    }
}

// ---------------------------------------------------------------------------
extern "C" void kernel_launch(void* const* d_in, const int* in_sizes, int n_in,
                              void* d_out, int out_size, void* d_ws, size_t ws_size,
                              hipStream_t stream) {
  const float* x  = (const float*)d_in[0];
  const float* Wa = (const float*)d_in[1];
  const float* ba = (const float*)d_in[2];
  const float* Wp = (const float*)d_in[3];
  const float* bp = (const float*)d_in[4];
  float* out = (float*)d_out;

  // Workspace layout (bf16 staging), total ~92 MB.
  char* ws = (char*)d_ws;
  size_t off = 0;
  __bf16* xb   = (__bf16*)(ws + off); off += (size_t)Mtot * Kdim * 2;  // 16 MB
  __bf16* wab  = (__bf16*)(ws + off); off += (size_t)Nqkv * Kdim * 2;  //  6 MB
  __bf16* wpb  = (__bf16*)(ws + off); off += (size_t)Cc   * Kdim * 2;  //  2 MB
  __bf16* qkvb = (__bf16*)(ws + off); off += (size_t)Mtot * Nqkv * 2;  // 48 MB
  __bf16* attb = (__bf16*)(ws + off);                                  // 16 MB

  auto cvt = [&](const float* s, __bf16* d, int n) {
    int threads = n / 8;
    cvt_f32_to_bf16<<<(threads + 255) / 256, 256, 0, stream>>>(s, d, n);
  };
  cvt(x,  xb,  Mtot * Kdim);
  cvt(Wa, wab, Nqkv * Kdim);
  cvt(Wp, wpb, Cc * Kdim);

  dim3 gq(Mtot / 128, Nqkv / 128);
  gemm_bf16_tn<Nqkv, true><<<gq, 256, 0, stream>>>(xb, wab, ba, (void*)qkvb);

  dim3 ga(Bb * Hh, Tt / 128);
  attn_flash<<<ga, 256, 0, stream>>>(qkvb, attb);

  dim3 gp(Mtot / 128, Cc / 128);
  gemm_bf16_tn<Cc, false><<<gp, 256, 0, stream>>>(attb, wpb, bp, (void*)out);
}
